// KlausLM_16578573762687
// MI455X (gfx1250) — matrix operand order repaired and checked by reading, not compiled, not run on hardware
//
#include <hip/hip_runtime.h>
#include <math.h>

// ---------------- model constants ----------------
constexpr int LAYERS = 6;
constexpr int DMODEL = 384;
constexpr int NHEAD  = 6;
constexpr int HEADD  = 64;
constexpr int HFF    = 768;
constexpr int VOCAB  = 4096;
constexpr int BATCH  = 4;
constexpr int SEQ    = 2048;
constexpr int MROWS  = BATCH * SEQ;   // 8192 token rows

// ---------------- types ----------------
typedef __attribute__((ext_vector_type(16))) __bf16        v16bf;
typedef __attribute__((ext_vector_type(8)))  float         v8f;
typedef __attribute__((ext_vector_type(8)))  unsigned int  v8u;
typedef __attribute__((ext_vector_type(4)))  int           v4i;

// CDNA5 async global->LDS copies (ASYNCcnt-tracked). Guarded so the host pass
// (and any toolchain lacking the builtin) falls back to synchronous staging.
#if __has_builtin(__builtin_amdgcn_global_load_async_to_lds_b128)
#define HAS_ASYNC_LDS 1
#else
#define HAS_ASYNC_LDS 0
#endif

#if HAS_ASYNC_LDS
__device__ __forceinline__ void async_b128(const unsigned short* g, unsigned short* l) {
  __builtin_amdgcn_global_load_async_to_lds_b128(
      (__attribute__((address_space(1))) v4i*)(unsigned short*)g,
      (__attribute__((address_space(3))) v4i*)l, 0, 0);
}
__device__ __forceinline__ void wait_async0() {
#if __has_builtin(__builtin_amdgcn_s_wait_asynccnt)
  __builtin_amdgcn_s_wait_asynccnt(0);
#else
  asm volatile("s_wait_asynccnt 0x0" ::: "memory");
#endif
}
#endif

__device__ __forceinline__ unsigned short f2bf(float f) {
  unsigned int u = __builtin_bit_cast(unsigned int, f);
  u = (u + 0x7fffu + ((u >> 16) & 1u)) >> 16;   // round-to-nearest-even
  return (unsigned short)u;
}
__device__ __forceinline__ v8f v8f_zero() {
  v8f z = {0.f, 0.f, 0.f, 0.f, 0.f, 0.f, 0.f, 0.f};
  return z;
}

// A-matrix fragment (16x32 bf16). Lane holds row m = lane%16; per ISA layout
// VGPRs 0-3 hold K = h*8 + 0..7, VGPRs 4-7 hold K = 16 + h*8 + 0..7 (h = lane/16).
// `row` points at 32 contiguous bf16 (K) values, 16B aligned (LDS or global).
__device__ __forceinline__ v16bf frag_a(const unsigned short* row, int hlf) {
  uint4 a = *(const uint4*)(row + hlf * 8);
  uint4 b = *(const uint4*)(row + 16 + hlf * 8);
  v8u t;
  t[0] = a.x; t[1] = a.y; t[2] = a.z; t[3] = a.w;
  t[4] = b.x; t[5] = b.y; t[6] = b.z; t[7] = b.w;
  return __builtin_bit_cast(v16bf, t);
}
// B-matrix fragment (32x16 bf16). Lane holds column n = lane%16; VGPRs 0-3 hold
// K = h*8 + 0..7, VGPRs 4-7 hold K = 16 + h*8 + 0..7 (h = lane/16), matching A.
__device__ __forceinline__ v16bf frag_b(const unsigned short* row, int hlf) {
  uint4 a = *(const uint4*)(row + hlf * 8);
  uint4 b = *(const uint4*)(row + 16 + hlf * 8);
  v8u t;
  t[0] = a.x; t[1] = a.y; t[2] = a.z; t[3] = a.w;
  t[4] = b.x; t[5] = b.y; t[6] = b.z; t[7] = b.w;
  return __builtin_bit_cast(v16bf, t);
}

// ---------------- weight convert + transpose: f32 [K][N] -> bf16 [N][K] ----------------
__global__ void wcvt_kernel(const float* __restrict__ w, unsigned short* __restrict__ wt,
                            int Kd, int Nd, float scale) {
  size_t i = (size_t)blockIdx.x * blockDim.x + threadIdx.x;
  if (i >= (size_t)Kd * Nd) return;
  int n = (int)(i % Nd);
  int kk = (int)(i / Nd);
  wt[(size_t)n * Kd + kk] = f2bf(w[i] * scale);
}

// ---------------- embedding gather ----------------
__global__ void embed_kernel(const int* __restrict__ x, const float* __restrict__ emb,
                             float* __restrict__ h) {
  int i = blockIdx.x * blockDim.x + threadIdx.x;     // over MROWS*DMODEL
  int row = i / DMODEL, col = i % DMODEL;
  h[i] = emb[(size_t)x[row] * DMODEL + col];
}

// ---------------- RMSNorm -> bf16 (consumed only as GEMM A operand) ----------------
__global__ __launch_bounds__(128) void rmsnorm_kernel(const float* __restrict__ x,
                                                      const float* __restrict__ w,
                                                      unsigned short* __restrict__ out) {
  __shared__ float red[128];
  int row = blockIdx.x, tid = threadIdx.x;
  const float* xr = x + (size_t)row * DMODEL;
  float ss = 0.f;
  for (int c = tid; c < DMODEL; c += 128) { float f = xr[c]; ss += f * f; }
  red[tid] = ss;
  __syncthreads();
  for (int s = 64; s > 0; s >>= 1) {
    if (tid < s) red[tid] += red[tid + s];
    __syncthreads();
  }
  float inv = rsqrtf(red[0] / (float)DMODEL + 1e-6f);
  for (int c = tid; c < DMODEL; c += 128)
    out[(size_t)row * DMODEL + c] = f2bf(xr[c] * inv * w[c]);
}

// ---------------- copy (for mid output) ----------------
__global__ void copy_kernel(const float* __restrict__ src, float* __restrict__ dst, int n) {
  int i = blockIdx.x * blockDim.x + threadIdx.x;
  if (i < n) dst[i] = src[i];
}

#if HAS_ASYNC_LDS
// Stage one 128x32 A tile + 64x32 B tile via async global->LDS b128 copies.
__device__ __forceinline__ void gemm_issue_tile(const unsigned short* A,
                                                const unsigned short* Bt,
                                                unsigned short* As, unsigned short* Bs,
                                                int m0, int n0, int K, int k0, int tid) {
#pragma unroll
  for (int c = 0; c < 4; c++)
    async_b128(A + (size_t)(m0 + tid) * K + k0 + c * 8, As + tid * 32 + c * 8);
  int n = tid >> 1, kk = (tid & 1) * 16;
#pragma unroll
  for (int c = 0; c < 2; c++)
    async_b128(Bt + (size_t)(n0 + n) * K + k0 + kk + c * 8, Bs + n * 32 + kk + c * 8);
}
#endif

// ---------------- tiled WMMA GEMM: out[M,N] = act(A @ Bt^T + bias) + res ----------------
// A: bf16 [M,K] row-major.  Bt: bf16 [N,K] row-major (pre-transposed weights).
// Block tile 128x64, BK=32; 4 waves, each 64x32 (4x2 tiles of 16x16).
// VTOUT: scatter bf16 output into [B][NH][HD][SEQ] (V for attention).
template <bool GELU, bool BF16OUT, bool VTOUT>
__global__ __launch_bounds__(128) void gemm_kernel(const unsigned short* __restrict__ A,
                                                   const unsigned short* __restrict__ Bt,
                                                   const float* __restrict__ bias,
                                                   const float* __restrict__ res,
                                                   void* __restrict__ out,
                                                   int M, int N, int K) {
  const int m0 = blockIdx.y * 128, n0 = blockIdx.x * 64;
  const int tid = threadIdx.x;
  const int lane = tid & 31, wave = tid >> 5;
  const int l16 = lane & 15, hlf = lane >> 4;
  const int mw = (wave & 1) * 64, nw = (wave >> 1) * 32;

  v8f acc[4][2];
#pragma unroll
  for (int mt = 0; mt < 4; mt++)
#pragma unroll
    for (int nt = 0; nt < 2; nt++) acc[mt][nt] = v8f_zero();

#if HAS_ASYNC_LDS
  // double-buffered: async copies for tile k+1 overlap WMMA on tile k
  __shared__ __align__(16) unsigned short As[2][128 * 32];
  __shared__ __align__(16) unsigned short Bs[2][64 * 32];
  gemm_issue_tile(A, Bt, As[0], Bs[0], m0, n0, K, 0, tid);
  wait_async0();
  __syncthreads();
  int cur = 0;
  for (int k0 = 0; k0 < K; k0 += 32) {
    if (k0 + 32 < K)
      gemm_issue_tile(A, Bt, As[cur ^ 1], Bs[cur ^ 1], m0, n0, K, k0 + 32, tid);
    v16bf af[4], bfm[2];
#pragma unroll
    for (int mt = 0; mt < 4; mt++)
      af[mt] = frag_a(As[cur] + (mw + mt * 16 + l16) * 32, hlf);
#pragma unroll
    for (int nt = 0; nt < 2; nt++)
      bfm[nt] = frag_b(Bs[cur] + (nw + nt * 16 + l16) * 32, hlf);
#pragma unroll
    for (int mt = 0; mt < 4; mt++)
#pragma unroll
      for (int nt = 0; nt < 2; nt++)
        acc[mt][nt] = __builtin_amdgcn_wmma_f32_16x16x32_bf16(
            false, af[mt], false, bfm[nt], (short)0, acc[mt][nt], false, false);
    wait_async0();
    __syncthreads();
    cur ^= 1;
  }
#else
  __shared__ __align__(16) unsigned short As[128 * 32];
  __shared__ __align__(16) unsigned short Bs[64 * 32];
  for (int k0 = 0; k0 < K; k0 += 32) {
    __syncthreads();
    {  // A tile: each thread copies one 32-element row (64 B)
      const uint4* src = (const uint4*)(A + (size_t)(m0 + tid) * K + k0);
      uint4* dst = (uint4*)(As + tid * 32);
      dst[0] = src[0]; dst[1] = src[1]; dst[2] = src[2]; dst[3] = src[3];
    }
    {  // B tile: 2 threads per N row (32 B each)
      int n = tid >> 1, kk = (tid & 1) * 16;
      const uint4* src = (const uint4*)(Bt + (size_t)(n0 + n) * K + k0 + kk);
      uint4* dst = (uint4*)(Bs + n * 32 + kk);
      dst[0] = src[0]; dst[1] = src[1];
    }
    if (k0 + 32 < K) {
      __builtin_prefetch(A + (size_t)(m0 + tid) * K + k0 + 32, 0, 1);
      __builtin_prefetch(Bt + (size_t)(n0 + (tid >> 1)) * K + k0 + 32, 0, 1);
    }
    __syncthreads();
    v16bf af[4], bfm[2];
#pragma unroll
    for (int mt = 0; mt < 4; mt++)
      af[mt] = frag_a(As + (mw + mt * 16 + l16) * 32, hlf);
#pragma unroll
    for (int nt = 0; nt < 2; nt++)
      bfm[nt] = frag_b(Bs + (nw + nt * 16 + l16) * 32, hlf);
#pragma unroll
    for (int mt = 0; mt < 4; mt++)
#pragma unroll
      for (int nt = 0; nt < 2; nt++)
        acc[mt][nt] = __builtin_amdgcn_wmma_f32_16x16x32_bf16(
            false, af[mt], false, bfm[nt], (short)0, acc[mt][nt], false, false);
  }
#endif

  // Epilogue. C layout: lane holds (N = l16, M = r + 8*hlf) in VGPR r.
#pragma unroll
  for (int mt = 0; mt < 4; mt++) {
#pragma unroll
    for (int nt = 0; nt < 2; nt++) {
      int gn = n0 + nw + nt * 16 + l16;
#pragma unroll
      for (int r = 0; r < 8; r++) {
        int gm = m0 + mw + mt * 16 + r + 8 * hlf;
        float val = acc[mt][nt][r];
        if (bias) val += bias[gn];
        if (GELU) val = 0.5f * val * (1.0f + erff(val * 0.70710678118654752f));
        if (res) val += res[(size_t)gm * N + gn];
        if constexpr (VTOUT) {
          int bq = gm >> 11, tt = gm & (SEQ - 1);         // SEQ = 2048
          int hh = gn >> 6, hd = gn & (HEADD - 1);
          ((unsigned short*)out)[(((size_t)bq * NHEAD + hh) * HEADD + hd) * SEQ + tt] =
              f2bf(val);
        } else if constexpr (BF16OUT) {
          ((unsigned short*)out)[(size_t)gm * N + gn] = f2bf(val);
        } else {
          ((float*)out)[(size_t)gm * N + gn] = val;
        }
      }
    }
  }
}

// ---------------- flash attention (causal), WMMA S and P@V ----------------
// q,k: bf16 [B*T][D] (scale 1/sqrt(HD) pre-folded into wq). vt: bf16 [B][NH][HD][T].
// grid (BATCH*NHEAD, SEQ/64), 128 threads. Wave w owns 16 query rows.
__global__ __launch_bounds__(128) void attn_kernel(const unsigned short* __restrict__ qbf,
                                                   const unsigned short* __restrict__ kbf,
                                                   const unsigned short* __restrict__ vtg,
                                                   unsigned short* __restrict__ op) {
  __shared__ __align__(16) unsigned short Ks[64 * 64];        // K rows [kt][hd]
  __shared__ __align__(16) unsigned short Vt[64 * 64];        // V^T rows [hd][kt]
  __shared__ __align__(16) unsigned short Pl[4][16 * 64];     // per-wave P [m][kt]

  const int bh = blockIdx.x;        // 0..B*NH-1
  const int qb = blockIdx.y;        // query block
  const int b = bh / NHEAD, hh = bh % NHEAD;
  const int tid = threadIdx.x, lane = tid & 31, w = tid >> 5;
  const int l16 = lane & 15, hlf = lane >> 4;

  // --- preload Q fragments (A layout) straight from bf16 global ---
  v16bf qf[2];
  {
    int qg = qb * 64 + w * 16 + l16;
    const unsigned short* qr = qbf + (size_t)(b * SEQ + qg) * DMODEL + hh * HEADD;
    qf[0] = frag_a(qr, hlf);
    qf[1] = frag_a(qr + 32, hlf);
  }

  float rm[8], rl[8];
  v8f oacc[4];
#pragma unroll
  for (int r = 0; r < 8; r++) { rm[r] = -1e30f; rl[r] = 0.f; }
#pragma unroll
  for (int nt = 0; nt < 4; nt++) oacc[nt] = v8f_zero();

  for (int kb = 0; kb <= qb; kb++) {
    __syncthreads();
#if HAS_ASYNC_LDS
    {  // K and V^T tiles: 512 x 16B chunks each, 4 per thread, pure byte copies
#pragma unroll
      for (int c = 0; c < 4; c++) {
        int ch = tid * 4 + c;
        int row = ch >> 3, off = (ch & 7) * 8;
        async_b128(kbf + (size_t)(b * SEQ + kb * 64 + row) * DMODEL + hh * HEADD + off,
                   Ks + row * 64 + off);
        async_b128(vtg + ((size_t)(b * NHEAD + hh) * HEADD + row) * SEQ + kb * 64 + off,
                   Vt + row * 64 + off);
      }
      wait_async0();
    }
#else
    {
      int row = tid >> 1, c0 = (tid & 1) * 32;
      const uint4* sk = (const uint4*)(kbf + (size_t)(b * SEQ + kb * 64 + row) * DMODEL +
                                       hh * HEADD + c0);
      uint4* dk = (uint4*)(Ks + row * 64 + c0);
      dk[0] = sk[0]; dk[1] = sk[1]; dk[2] = sk[2]; dk[3] = sk[3];
      const uint4* sv = (const uint4*)(vtg + ((size_t)(b * NHEAD + hh) * HEADD + row) * SEQ +
                                       kb * 64 + c0);
      uint4* dv = (uint4*)(Vt + row * 64 + c0);
      dv[0] = sv[0]; dv[1] = sv[1]; dv[2] = sv[2]; dv[3] = sv[3];
    }
#endif
    __syncthreads();

    // --- S = (Q/8) @ K^T : 4 tiles of 16x16, K-dim = 64 (2 WMMA steps) ---
    v8f s[4];
#pragma unroll
    for (int nt = 0; nt < 4; nt++) {
      v8f a = v8f_zero();
#pragma unroll
      for (int ks = 0; ks < 2; ks++) {
        v16bf kf = frag_b(Ks + (nt * 16 + l16) * 64 + ks * 32, hlf);
        a = __builtin_amdgcn_wmma_f32_16x16x32_bf16(false, qf[ks], false, kf,
                                                    (short)0, a, false, false);
      }
      s[nt] = a;
    }

    // --- causal mask (diagonal block only) + online softmax ---
    const bool diag = (kb == qb);
    float curmax[8];
#pragma unroll
    for (int r = 0; r < 8; r++) curmax[r] = -1e30f;
#pragma unroll
    for (int nt = 0; nt < 4; nt++) {
      int kg = kb * 64 + nt * 16 + l16;
#pragma unroll
      for (int r = 0; r < 8; r++) {
        int qg = qb * 64 + w * 16 + r + 8 * hlf;
        float val = s[nt][r];
        if (diag && kg > qg) val = -1e30f;
        s[nt][r] = val;
        curmax[r] = fmaxf(curmax[r], val);
      }
    }
#pragma unroll
    for (int r = 0; r < 8; r++) {  // row max across the 16-lane half
      float m_ = curmax[r];
      m_ = fmaxf(m_, __shfl_xor(m_, 1, 32));
      m_ = fmaxf(m_, __shfl_xor(m_, 2, 32));
      m_ = fmaxf(m_, __shfl_xor(m_, 4, 32));
      m_ = fmaxf(m_, __shfl_xor(m_, 8, 32));
      curmax[r] = m_;
    }
    float alpha[8];
#pragma unroll
    for (int r = 0; r < 8; r++) {
      float nm = fmaxf(rm[r], curmax[r]);
      alpha[r] = __expf(rm[r] - nm);
      rm[r] = nm;
    }
    float rsum[8];
#pragma unroll
    for (int r = 0; r < 8; r++) rsum[r] = 0.f;
    unsigned short* pw = Pl[w];
#pragma unroll
    for (int nt = 0; nt < 4; nt++) {
#pragma unroll
      for (int r = 0; r < 8; r++) {
        float pvl = __expf(s[nt][r] - rm[r]);
        rsum[r] += pvl;
        pw[(r + 8 * hlf) * 64 + nt * 16 + l16] = f2bf(pvl);
      }
    }
#pragma unroll
    for (int r = 0; r < 8; r++) {
      float t_ = rsum[r];
      t_ += __shfl_xor(t_, 1, 32);
      t_ += __shfl_xor(t_, 2, 32);
      t_ += __shfl_xor(t_, 4, 32);
      t_ += __shfl_xor(t_, 8, 32);
      rl[r] = rl[r] * alpha[r] + t_;
    }
#pragma unroll
    for (int nt = 0; nt < 4; nt++)
#pragma unroll
      for (int r = 0; r < 8; r++) oacc[nt][r] *= alpha[r];

    __syncthreads();  // P (C-layout) now in LDS; reload in A-layout

    // --- O += P @ V : 4 hd-tiles, K-dim = 64 kt (2 WMMA steps) ---
    v16bf pf[2];
#pragma unroll
    for (int ks = 0; ks < 2; ks++)
      pf[ks] = frag_a(pw + l16 * 64 + ks * 32, hlf);
#pragma unroll
    for (int nt = 0; nt < 4; nt++) {
#pragma unroll
      for (int ks = 0; ks < 2; ks++) {
        v16bf vf = frag_b(Vt + (nt * 16 + l16) * 64 + ks * 32, hlf);
        oacc[nt] = __builtin_amdgcn_wmma_f32_16x16x32_bf16(false, pf[ks], false, vf,
                                                           (short)0, oacc[nt], false, false);
      }
    }
  }

  // --- normalize and write bf16 output row-major [B*T, D] ---
#pragma unroll
  for (int nt = 0; nt < 4; nt++) {
#pragma unroll
    for (int r = 0; r < 8; r++) {
      int qg = qb * 64 + w * 16 + r + 8 * hlf;
      float val = oacc[nt][r] / rl[r];
      op[(size_t)(b * SEQ + qg) * DMODEL + hh * HEADD + nt * 16 + l16] = f2bf(val);
    }
  }
}

// ---------------- host orchestration ----------------
extern "C" void kernel_launch(void* const* d_in, const int* in_sizes, int n_in,
                              void* d_out, int out_size, void* d_ws, size_t ws_size,
                              hipStream_t stream) {
  (void)in_sizes; (void)n_in; (void)out_size; (void)ws_size;
  const int*   x    = (const int*)d_in[0];
  const float* emb  = (const float*)d_in[1];
  const float* wq   = (const float*)d_in[2];
  const float* wk   = (const float*)d_in[3];
  const float* wv   = (const float*)d_in[4];
  const float* wo   = (const float*)d_in[5];
  const float* ln1  = (const float*)d_in[6];
  const float* ln2  = (const float*)d_in[7];
  const float* w1   = (const float*)d_in[8];
  const float* b1   = (const float*)d_in[9];
  const float* w2   = (const float*)d_in[10];
  const float* b2   = (const float*)d_in[11];
  const float* lnf  = (const float*)d_in[12];
  const float* head = (const float*)d_in[13];

  float* logits = (float*)d_out;
  float* midout = logits + (size_t)BATCH * SEQ * VOCAB;

  // --- carve workspace ---
  char* p = (char*)d_ws;
  auto carve = [&](size_t bytes) -> void* {
    void* r = (void*)p;
    p += (bytes + 255) & ~(size_t)255;
    return r;
  };
  unsigned short* wqT = (unsigned short*)carve((size_t)LAYERS * DMODEL * DMODEL * 2);
  unsigned short* wkT = (unsigned short*)carve((size_t)LAYERS * DMODEL * DMODEL * 2);
  unsigned short* wvT = (unsigned short*)carve((size_t)LAYERS * DMODEL * DMODEL * 2);
  unsigned short* woT = (unsigned short*)carve((size_t)LAYERS * DMODEL * DMODEL * 2);
  unsigned short* w1T = (unsigned short*)carve((size_t)LAYERS * DMODEL * HFF * 2);
  unsigned short* w2T = (unsigned short*)carve((size_t)LAYERS * HFF * DMODEL * 2);
  unsigned short* hdT = (unsigned short*)carve((size_t)DMODEL * VOCAB * 2);
  float*          hbuf = (float*)carve((size_t)MROWS * DMODEL * 4);
  unsigned short* xnbf = (unsigned short*)carve((size_t)MROWS * DMODEL * 2);
  unsigned short* qbf  = (unsigned short*)carve((size_t)MROWS * DMODEL * 2);
  unsigned short* kbf  = (unsigned short*)carve((size_t)MROWS * DMODEL * 2);
  unsigned short* vtbf = (unsigned short*)carve((size_t)MROWS * DMODEL * 2);
  unsigned short* atbf = (unsigned short*)carve((size_t)MROWS * DMODEL * 2);
  unsigned short* ffbf = (unsigned short*)carve((size_t)MROWS * HFF * 2);

  auto cvtw = [&](const float* src, unsigned short* dst, int Kd, int Nd, float scale) {
    size_t n = (size_t)Kd * Nd;
    wcvt_kernel<<<dim3((unsigned)((n + 255) / 256)), 256, 0, stream>>>(src, dst, Kd, Nd,
                                                                       scale);
  };

  // --- weights: convert + transpose to bf16 [N][K]; fold 1/sqrt(HD) into wq ---
  for (int i = 0; i < LAYERS; i++) {
    size_t dd = (size_t)i * DMODEL * DMODEL;
    cvtw(wq + dd, wqT + dd, DMODEL, DMODEL, 0.125f);
    cvtw(wk + dd, wkT + dd, DMODEL, DMODEL, 1.0f);
    cvtw(wv + dd, wvT + dd, DMODEL, DMODEL, 1.0f);
    cvtw(wo + dd, woT + dd, DMODEL, DMODEL, 1.0f);
    cvtw(w1 + (size_t)i * DMODEL * HFF, w1T + (size_t)i * DMODEL * HFF, DMODEL, HFF, 1.0f);
    cvtw(w2 + (size_t)i * HFF * DMODEL, w2T + (size_t)i * HFF * DMODEL, HFF, DMODEL, 1.0f);
  }
  cvtw(head, hdT, DMODEL, VOCAB, 1.0f);

  // --- embedding ---
  embed_kernel<<<dim3(MROWS * DMODEL / 256), 256, 0, stream>>>(x, emb, hbuf);

  const dim3 gemmD(DMODEL / 64, MROWS / 128);   // N=384
  const dim3 gemmH(HFF / 64, MROWS / 128);      // N=768
  const dim3 gemmV(VOCAB / 64, MROWS / 128);    // N=4096

  for (int i = 0; i < LAYERS; i++) {
    size_t dd = (size_t)i * DMODEL * DMODEL;
    // xn = rms(h, ln1)
    rmsnorm_kernel<<<dim3(MROWS), 128, 0, stream>>>(hbuf, ln1 + (size_t)i * DMODEL, xnbf);
    // q,k bf16 row-major; v bf16 transposed [B][NH][HD][T]
    gemm_kernel<false, true, false><<<gemmD, 128, 0, stream>>>(
        xnbf, wqT + dd, nullptr, nullptr, qbf, MROWS, DMODEL, DMODEL);
    gemm_kernel<false, true, false><<<gemmD, 128, 0, stream>>>(
        xnbf, wkT + dd, nullptr, nullptr, kbf, MROWS, DMODEL, DMODEL);
    gemm_kernel<false, true, true><<<gemmD, 128, 0, stream>>>(
        xnbf, wvT + dd, nullptr, nullptr, vtbf, MROWS, DMODEL, DMODEL);
    // causal attention
    attn_kernel<<<dim3(BATCH * NHEAD, SEQ / 64), 128, 0, stream>>>(qbf, kbf, vtbf, atbf);
    // h += attn @ wo
    gemm_kernel<false, false, false><<<gemmD, 128, 0, stream>>>(
        atbf, woT + dd, nullptr, hbuf, hbuf, MROWS, DMODEL, DMODEL);
    // xn = rms(h, ln2); ff = gelu(xn@w1 + b1); h += ff@w2 + b2
    rmsnorm_kernel<<<dim3(MROWS), 128, 0, stream>>>(hbuf, ln2 + (size_t)i * DMODEL, xnbf);
    gemm_kernel<true, true, false><<<gemmH, 128, 0, stream>>>(
        xnbf, w1T + (size_t)i * DMODEL * HFF, b1 + (size_t)i * HFF, nullptr, ffbf,
        MROWS, HFF, DMODEL);
    gemm_kernel<false, false, false><<<gemmD, 128, 0, stream>>>(
        ffbf, w2T + (size_t)i * HFF * DMODEL, b2 + (size_t)i * DMODEL, hbuf, hbuf,
        MROWS, DMODEL, HFF);
    if (i == 2)
      copy_kernel<<<dim3(MROWS * DMODEL / 256), 256, 0, stream>>>(hbuf, midout,
                                                                  MROWS * DMODEL);
  }

  // logits = rms(h, lnf) @ head
  rmsnorm_kernel<<<dim3(MROWS), 128, 0, stream>>>(hbuf, lnf, xnbf);
  gemm_kernel<false, false, false><<<gemmV, 128, 0, stream>>>(
      xnbf, hdT, nullptr, nullptr, logits, MROWS, VOCAB, DMODEL);
}
